// MultiHeadAttention_42004780155164
// MI455X (gfx1250) — compile-verified, matrix operand order checked by reading, not run on hardware
//
#include <hip/hip_runtime.h>
#include <hip/hip_bf16.h>

typedef __attribute__((ext_vector_type(16))) _Float16 v16h;
typedef __attribute__((ext_vector_type(8)))  float    v8f;

#define B_    8
#define N_    1024
#define FEAT_ 1024
#define HID_  1024
#define OUT_  1024
#define SDIM_ 512
#define H_    16
#define HD_   64

// ---------------- CDNA5 async global->LDS copy (ASYNCcnt-tracked) ----------------

__device__ __forceinline__ void async_copy_b128(void* lds_dst, const void* gsrc) {
  // LDS flat address: low 32 bits are the wave-relative LDS byte offset.
  const unsigned off = (unsigned)(unsigned long long)lds_dst;
  asm volatile("global_load_async_to_lds_b128 %0, %1, off"
               :
               : "v"(off), "v"(gsrc)
               : "memory");
}

__device__ __forceinline__ void wait_async() {
  asm volatile("s_wait_asynccnt 0" ::: "memory");
}

// ---------------- WMMA fragment helpers (gfx1250 wave32 layouts) ----------------

__device__ __forceinline__ v8f wmma_f16(v16h a, v16h b, v8f c) {
  // v_wmma_f32_16x16x32_f16: (neg_a, A, neg_b, B, c_mod, C, reuse_a, reuse_b)
  return __builtin_amdgcn_wmma_f32_16x16x32_f16(false, a, false, b, (short)0, c,
                                                false, false);
}

// A fragment (16x32 f16): lane = M row (lane&15); K-block from lane half.
__device__ __forceinline__ v16h frag_a_load(const _Float16* src, int ld) {
  const int lane = threadIdx.x & 31;
  const int row  = lane & 15;
  const int kb   = (lane >> 4) << 3;  // 0 or 8
  v16h f;
#pragma unroll
  for (int i = 0; i < 4; ++i) {
    const int k  = kb + 2 * i;
    const int k2 = kb + 16 + 2 * i;
    f[2 * i]           = src[row * ld + k];
    f[2 * i + 1]       = src[row * ld + k + 1];
    f[2 * (i + 4)]     = src[row * ld + k2];
    f[2 * (i + 4) + 1] = src[row * ld + k2 + 1];
  }
  return f;
}

// B fragment (32x16 f16) from row-major B[k][n] (stride ld, n contiguous).
__device__ __forceinline__ v16h frag_b_load(const _Float16* src, int ld) {
  const int lane = threadIdx.x & 31;
  const int col  = lane & 15;
  const int kb   = (lane >> 4) << 3;
  v16h f;
#pragma unroll
  for (int i = 0; i < 4; ++i) {
    const int k  = kb + 2 * i;
    const int k2 = kb + 16 + 2 * i;
    f[2 * i]           = src[k * ld + col];
    f[2 * i + 1]       = src[(k + 1) * ld + col];
    f[2 * (i + 4)]     = src[k2 * ld + col];
    f[2 * (i + 4) + 1] = src[(k2 + 1) * ld + col];
  }
  return f;
}

// B fragment where B = src^T, src stored row-major [n][k] (stride ld).
__device__ __forceinline__ v16h frag_bt_load(const _Float16* src, int ld) {
  const int lane = threadIdx.x & 31;
  const int col  = lane & 15;
  const int kb   = (lane >> 4) << 3;
  v16h f;
#pragma unroll
  for (int i = 0; i < 4; ++i) {
    f[2 * i]           = src[col * ld + kb + 2 * i];
    f[2 * i + 1]       = src[col * ld + kb + 2 * i + 1];
    f[2 * (i + 4)]     = src[col * ld + kb + 16 + 2 * i];
    f[2 * (i + 4) + 1] = src[col * ld + kb + 16 + 2 * i + 1];
  }
  return f;
}

// ---------------- Stage 0: style vectors + demod factors (tiny, VALU) ----------------

__global__ __launch_bounds__(256) void style_kernel(
    const float* __restrict__ s, const float* __restrict__ aw,
    const float* __restrict__ ab, float* __restrict__ style,
    int Cin, int Sdim) {
  const int idx = blockIdx.x * blockDim.x + threadIdx.x;
  if (idx >= B_ * Cin) return;
  const int b = idx / Cin, i = idx - b * Cin;
  const float* sv = s + (size_t)b * Sdim;
  const float* wv = aw + (size_t)i * Sdim;
  float acc = 0.f;
  for (int j = 0; j < Sdim; ++j) acc += sv[j] * wv[j];
  style[idx] = acc + ab[i];
}

__global__ __launch_bounds__(256) void demod_kernel(
    const float* __restrict__ w, const float* __restrict__ style,
    float* __restrict__ demod, int Cout, int Cin) {
  const int idx = blockIdx.x * blockDim.x + threadIdx.x;
  if (idx >= B_ * Cout) return;
  const int b = idx / Cout, o = idx - b * Cout;
  const float* st = style + (size_t)b * Cin;
  const float* wv = w + (size_t)o * Cin;
  float acc = 0.f;
  for (int i = 0; i < Cin; ++i) { const float t = wv[i] * st[i]; acc += t * t; }
  demod[idx] = rsqrtf(acc + 1e-8f);
}

// ---------------- Modulated GEMM: out = (x*style) @ w^T * demod + bias ----------------
// MODE 0: x is f32, result -> f16 head layout [B,H,N,HD]
// MODE 1: x is f16 (attention output), result -> f32 [B,N,Cout]
// 64x64 tile per 128-thread block, K-step 64 (8 wmma per wave per barrier).

template <int MODE>
__global__ __launch_bounds__(128) void modlin_kernel(
    const float* __restrict__ xf32, const _Float16* __restrict__ xf16,
    const float* __restrict__ w, const float* __restrict__ bias,
    const float* __restrict__ style, const float* __restrict__ demod,
    _Float16* __restrict__ out16, float* __restrict__ outf32) {
  constexpr int Cin = 1024, Cout = 1024;
  const int m0 = blockIdx.x * 64;
  const int n0 = blockIdx.y * 64;
  const int b  = blockIdx.z;

  __shared__ _Float16 As[64 * 72];  // 64 x 64 tile, ld=72 (pad)
  __shared__ _Float16 Bs[64 * 72];  // 64 x 64 tile, ld=72 (pad)

  const int tid  = threadIdx.x;
  const int wave = tid >> 5;
  const int wm = (wave >> 1) * 32;  // wave's 32x32 quadrant
  const int wn = (wave & 1) * 32;

  v8f acc[2][2] = {};

  const float* stl = style + (size_t)b * Cin;
  for (int k0 = 0; k0 < Cin; k0 += 64) {
    for (int e = tid; e < 64 * 64; e += 128) {
      const int r = e >> 6, k = e & 63;
      float v;
      if (MODE == 0)
        v = xf32[((size_t)b * N_ + (m0 + r)) * Cin + k0 + k] * stl[k0 + k];
      else
        v = (float)xf16[((size_t)b * N_ + (m0 + r)) * Cin + k0 + k] * stl[k0 + k];
      As[r * 72 + k] = (_Float16)v;
    }
    for (int e = tid; e < 64 * 64; e += 128) {
      const int k = e >> 6, n = e & 63;
      Bs[k * 72 + n] = (_Float16)w[(size_t)(n0 + n) * Cin + k0 + k];
    }
    __syncthreads();

#pragma unroll
    for (int kc = 0; kc < 2; ++kc) {
      const v16h a0 = frag_a_load(&As[(wm + 0) * 72 + 32 * kc], 72);
      const v16h a1 = frag_a_load(&As[(wm + 16) * 72 + 32 * kc], 72);
      const v16h b0 = frag_b_load(&Bs[(32 * kc) * 72 + wn + 0], 72);
      const v16h b1 = frag_b_load(&Bs[(32 * kc) * 72 + wn + 16], 72);
      acc[0][0] = wmma_f16(a0, b0, acc[0][0]);
      acc[0][1] = wmma_f16(a0, b1, acc[0][1]);
      acc[1][0] = wmma_f16(a1, b0, acc[1][0]);
      acc[1][1] = wmma_f16(a1, b1, acc[1][1]);
    }
    __syncthreads();
  }

  // Epilogue: demod scale + bias. C layout: lane -> col = lane&15, rows = (lane>>4)*8 + r
  const int lane = tid & 31;
  const int colL = lane & 15;
  const int rowB = (lane >> 4) * 8;
  const float* dm = demod + (size_t)b * Cout;
#pragma unroll
  for (int i = 0; i < 2; ++i)
#pragma unroll
    for (int j = 0; j < 2; ++j) {
      const int cg = n0 + wn + 16 * j + colL;
      const float d  = dm[cg];
      const float bb = bias[cg];
#pragma unroll
      for (int r = 0; r < 8; ++r) {
        const int rg = m0 + wm + 16 * i + rowB + r;
        const float v = acc[i][j][r] * d + bb;
        if (MODE == 0) {
          const int head = cg >> 6, dd = cg & 63;
          out16[(((size_t)b * H_ + head) * N_ + rg) * HD_ + dd] = (_Float16)v;
        } else {
          outf32[((size_t)b * N_ + rg) * Cout + cg] = v;
        }
      }
    }
}

// ---------------- Flash attention, q=k=v=h, per (b, head, 64-row block) ----------------
// K and V tiles are the same data (self-attention on h), so one LDS tile serves both.
// Tile staging uses CDNA5 async global->LDS B128 copies (ASYNCcnt).

__global__ __launch_bounds__(128) void attn_kernel(
    const _Float16* __restrict__ h16,  // [B,H,N,HD]
    _Float16* __restrict__ attn16) {   // [B,N,HID] f16
  const int q0 = blockIdx.x * 64;
  const int h  = blockIdx.y;
  const int b  = blockIdx.z;
  const _Float16* Hm = h16 + ((size_t)b * H_ + h) * N_ * HD_;

  __shared__ _Float16 Qs[64 * 72];
  __shared__ _Float16 Ks[64 * 72];       // serves as both K and V tile
  __shared__ _Float16 Ps[4][16 * 72];

  const int tid  = threadIdx.x;
  const int wave = tid >> 5;
  const int lane = tid & 31;
  const int wq   = wave * 16;  // this wave's 16 query rows within the 64 block

  // Stage Q tile: 64 rows x 64 halves = 512 b128 chunks, async copies.
  for (int e = tid; e < 64 * 8; e += 128) {
    const int r = e >> 3, c = e & 7;  // c indexes 8-half (16B) chunks
    async_copy_b128(&Qs[r * 72 + c * 8], Hm + (size_t)(q0 + r) * HD_ + c * 8);
  }
  wait_async();
  __syncthreads();

  const v16h qa0 = frag_a_load(&Qs[wq * 72 + 0], 72);
  const v16h qa1 = frag_a_load(&Qs[wq * 72 + 32], 72);

  v8f o[4] = {};
  float m_[8], l_[8];
#pragma unroll
  for (int j = 0; j < 8; ++j) { m_[j] = -1e30f; l_[j] = 0.f; }

  const float scl = 0.125f;  // 1/sqrt(64)
  for (int kv0 = 0; kv0 < N_; kv0 += 64) {
    // Stage K/V tile (shared data) via async copies.
    for (int e = tid; e < 64 * 8; e += 128) {
      const int r = e >> 3, c = e & 7;
      async_copy_b128(&Ks[r * 72 + c * 8], Hm + (size_t)(kv0 + r) * HD_ + c * 8);
    }
    wait_async();
    __syncthreads();

    // S = Q @ K^T  (4 col-fragments of 16 kv each, K=64 in two 32-chunks)
    v8f s[4];
#pragma unroll
    for (int jn = 0; jn < 4; ++jn) {
      v8f z = {};
      const v16h kb0 = frag_bt_load(&Ks[(16 * jn) * 72 + 0], 72);
      const v16h kb1 = frag_bt_load(&Ks[(16 * jn) * 72 + 32], 72);
      z = wmma_f16(qa0, kb0, z);
      z = wmma_f16(qa1, kb1, z);
      s[jn] = z;
    }

    // Online softmax: row stats replicated across each 16-lane half.
    float rmax[8];
#pragma unroll
    for (int j = 0; j < 8; ++j) {
      float mx = fmaxf(fmaxf(s[0][j], s[1][j]), fmaxf(s[2][j], s[3][j]));
      rmax[j] = mx * scl;
    }
#pragma unroll
    for (int j = 0; j < 8; ++j) {
#pragma unroll
      for (int msk = 1; msk < 16; msk <<= 1)
        rmax[j] = fmaxf(rmax[j], __shfl_xor(rmax[j], msk, 32));
    }
#pragma unroll
    for (int j = 0; j < 8; ++j) {
      const float mnew = fmaxf(m_[j], rmax[j]);
      const float corr = __expf(m_[j] - mnew);
      float ps = 0.f;
#pragma unroll
      for (int jn = 0; jn < 4; ++jn) {
        const float p = __expf(s[jn][j] * scl - mnew);
        s[jn][j] = p;
        ps += p;
      }
#pragma unroll
      for (int msk = 1; msk < 16; msk <<= 1) ps += __shfl_xor(ps, msk, 32);
      l_[j] = l_[j] * corr + ps;
      m_[j] = mnew;
#pragma unroll
      for (int jd = 0; jd < 4; ++jd) o[jd][j] *= corr;
    }

    // C-layout -> A-layout for P via per-wave LDS region (same-wave DS is in-order).
    const int colL = lane & 15;
    const int rowB = (lane >> 4) * 8;
#pragma unroll
    for (int jn = 0; jn < 4; ++jn)
#pragma unroll
      for (int j = 0; j < 8; ++j)
        Ps[wave][(rowB + j) * 72 + 16 * jn + colL] = (_Float16)s[jn][j];

    const v16h pa0 = frag_a_load(&Ps[wave][0], 72);
    const v16h pa1 = frag_a_load(&Ps[wave][32], 72);
#pragma unroll
    for (int jd = 0; jd < 4; ++jd) {
      // V == K tile: B[k=kv][n=dim] is just the row-major tile.
      const v16h vb0 = frag_b_load(&Ks[0 * 72 + 16 * jd], 72);
      const v16h vb1 = frag_b_load(&Ks[32 * 72 + 16 * jd], 72);
      o[jd] = wmma_f16(pa0, vb0, o[jd]);
      o[jd] = wmma_f16(pa1, vb1, o[jd]);
    }
    __syncthreads();
  }

  // Normalize and store to [B,N,HID] f16
  const int colL = lane & 15;
  const int rowB = (lane >> 4) * 8;
#pragma unroll
  for (int j = 0; j < 8; ++j) {
    const float inv = 1.f / l_[j];
    const int rg = q0 + wq + rowB + j;
#pragma unroll
    for (int jd = 0; jd < 4; ++jd) {
      const int dd = 16 * jd + colL;
      attn16[((size_t)b * N_ + rg) * HID_ + h * 64 + dd] =
          (_Float16)(o[jd][j] * inv);
    }
  }
}

// ---------------- Launch ----------------

extern "C" void kernel_launch(void* const* d_in, const int* in_sizes, int n_in,
                              void* d_out, int out_size, void* d_ws, size_t ws_size,
                              hipStream_t stream) {
  (void)in_sizes; (void)n_in; (void)out_size; (void)ws_size;
  const float* x   = (const float*)d_in[0];
  const float* s   = (const float*)d_in[1];
  const float* kaw = (const float*)d_in[2];
  const float* kab = (const float*)d_in[3];
  const float* kw  = (const float*)d_in[4];
  const float* kb  = (const float*)d_in[5];
  const float* oaw = (const float*)d_in[6];
  const float* oab = (const float*)d_in[7];
  const float* ow  = (const float*)d_in[8];
  const float* ob  = (const float*)d_in[9];
  float* out = (float*)d_out;

  char* ws = (char*)d_ws;
  float* style_k = (float*)ws;  ws += (size_t)B_ * FEAT_ * sizeof(float);
  float* style_o = (float*)ws;  ws += (size_t)B_ * HID_ * sizeof(float);
  float* demod_k = (float*)ws;  ws += (size_t)B_ * HID_ * sizeof(float);
  float* demod_o = (float*)ws;  ws += (size_t)B_ * OUT_ * sizeof(float);
  _Float16* h16    = (_Float16*)ws;  ws += (size_t)B_ * H_ * N_ * HD_ * sizeof(_Float16);
  _Float16* attn16 = (_Float16*)ws;

  style_kernel<<<(B_ * FEAT_ + 255) / 256, 256, 0, stream>>>(s, kaw, kab, style_k, FEAT_, SDIM_);
  style_kernel<<<(B_ * HID_ + 255) / 256, 256, 0, stream>>>(s, oaw, oab, style_o, HID_, SDIM_);
  demod_kernel<<<(B_ * HID_ + 255) / 256, 256, 0, stream>>>(kw, style_k, demod_k, HID_, FEAT_);
  demod_kernel<<<(B_ * OUT_ + 255) / 256, 256, 0, stream>>>(ow, style_o, demod_o, OUT_, HID_);

  dim3 g1(N_ / 64, HID_ / 64, B_);
  modlin_kernel<0><<<g1, 128, 0, stream>>>(x, nullptr, kw, kb, style_k, demod_k, h16, nullptr);

  dim3 ga(N_ / 64, H_, B_);
  attn_kernel<<<ga, 128, 0, stream>>>(h16, attn16);

  dim3 g2(N_ / 64, OUT_ / 64, B_);
  modlin_kernel<1><<<g2, 128, 0, stream>>>(nullptr, attn16, ow, ob, style_o, demod_o, nullptr, out);
}
